// IwinTransformer3D_47425028882596
// MI455X (gfx1250) — matrix operand order repaired and checked by reading, not compiled, not run on hardware
//
#include <hip/hip_runtime.h>
#include <math.h>

typedef __bf16 bf16;
typedef __attribute__((ext_vector_type(16))) __bf16 v16bf;
typedef __attribute__((ext_vector_type(4)))  __bf16 v4bf;
typedef __attribute__((ext_vector_type(8)))  float  v8f;

#define NTOK   50176      // B*D*H*W = 2*8*56*56
#define CDIM   128
#define NHEAD  4
#define HD     32
#define NWIN   128        // B * 8 * 8
#define NTOKW  392        // D*7*7 tokens per window
#define HIDDEN 512

// ---------------------------------------------------------------------------
// LayerNorm: one wave per token (C=128 -> float4 per lane), shuffle reduce.
// ---------------------------------------------------------------------------
__global__ __launch_bounds__(256) void ln_kernel(
    const float* __restrict__ x, const float* __restrict__ g,
    const float* __restrict__ b, bf16* __restrict__ out_bf,
    float* __restrict__ out_f, int ntok)
{
  int wv = threadIdx.x >> 5, lane = threadIdx.x & 31;
  int t = blockIdx.x * 8 + wv;
  if (t >= ntok) return;
  const float* xp = x + (size_t)t * CDIM;
  float4 v = reinterpret_cast<const float4*>(xp)[lane];
  float s  = v.x + v.y + v.z + v.w;
  float sq = v.x*v.x + v.y*v.y + v.z*v.z + v.w*v.w;
  #pragma unroll
  for (int m = 1; m < 32; m <<= 1) {
    s  += __shfl_xor(s,  m, 32);
    sq += __shfl_xor(sq, m, 32);
  }
  float mu  = s * (1.0f / CDIM);
  float var = sq * (1.0f / CDIM) - mu * mu;
  float rs  = rsqrtf(var + 1e-5f);
  float4 gg = reinterpret_cast<const float4*>(g)[lane];
  float4 bb = reinterpret_cast<const float4*>(b)[lane];
  float y0 = (v.x - mu) * rs * gg.x + bb.x;
  float y1 = (v.y - mu) * rs * gg.y + bb.y;
  float y2 = (v.z - mu) * rs * gg.z + bb.z;
  float y3 = (v.w - mu) * rs * gg.w + bb.w;
  size_t base = (size_t)t * CDIM + lane * 4;
  v4bf pk; pk[0] = (bf16)y0; pk[1] = (bf16)y1; pk[2] = (bf16)y2; pk[3] = (bf16)y3;
  *reinterpret_cast<v4bf*>(out_bf + base) = pk;
  if (out_f) {
    float4 o; o.x = y0; o.y = y1; o.z = y2; o.w = y3;
    reinterpret_cast<float4*>(out_f + (size_t)t * CDIM)[lane] = o;
  }
}

// ---------------------------------------------------------------------------
// Depthwise 3x3 SAME conv per frame (16 frames of 56x56x128). Pure VALU,
// float4 (4 channels) per thread, coalesced on channel.
// ---------------------------------------------------------------------------
__global__ __launch_bounds__(256) void dwconv_kernel(
    const float* __restrict__ xn, const float* __restrict__ dw_w,
    const float* __restrict__ dw_b, float* __restrict__ out)
{
  int gid = blockIdx.x * 256 + threadIdx.x;     // token*32 + c4
  int c4 = gid & 31;
  int t  = gid >> 5;
  int w = t % 56;
  int h = (t / 56) % 56;
  int f = t / (56 * 56);
  float4 acc = reinterpret_cast<const float4*>(dw_b)[c4];
  #pragma unroll
  for (int kh = 0; kh < 3; ++kh) {
    #pragma unroll
    for (int kw = 0; kw < 3; ++kw) {
      int hh = h + kh - 1, ww = w + kw - 1;
      if (hh >= 0 && hh < 56 && ww >= 0 && ww < 56) {
        float4 xv = reinterpret_cast<const float4*>(
            xn + (((size_t)f * 56 + hh) * 56 + ww) * CDIM)[c4];
        float4 wv = reinterpret_cast<const float4*>(
            dw_w + (kh * 3 + kw) * CDIM)[c4];
        acc.x += xv.x * wv.x; acc.y += xv.y * wv.y;
        acc.z += xv.z * wv.z; acc.w += xv.w * wv.w;
      }
    }
  }
  reinterpret_cast<float4*>(out + (size_t)t * CDIM)[c4] = acc;
}

// ---------------------------------------------------------------------------
// Generic bf16 WMMA GEMM: out[M,N] = A[M,K](bf16) * W[N,K]^T(f32) + bias.
// Block tile 128x64, K staged in 128-deep chunks (~51 KB LDS), 16 WMMAs per
// barrier pair. 8 waves: wave w -> rows w*16..w*16+15, 4 col tiles each.
// LDS rows padded (+8 bf16) to spread banks; all fragment reads contiguous.
// EPI: 0 = bias -> bf16             (qkv)
//      1 = bias, exact GELU -> bf16 (fc1)
//      2 = bias + aux0 + aux1 -> f32 (proj: +conv +shortcut -> x1)
//      3 = bias + aux0 -> f32        (fc2: +x1 -> final out)
// ---------------------------------------------------------------------------
#define APAD 136   // 128 + 8
template <int EPI>
__global__ __launch_bounds__(256) void gemm_kernel(
    const bf16* __restrict__ A, const float* __restrict__ W,
    const float* __restrict__ bias, const float* __restrict__ aux0,
    const float* __restrict__ aux1, void* __restrict__ outv,
    int M, int N, int K)
{
  __shared__ bf16 As[128][APAD];   // ~34 KB
  __shared__ bf16 Bst[64][APAD];   // ~17 KB, Bst[n][k] = W[n0+n][kc+k]
  const int tid = threadIdx.x;
  const int wv = tid >> 5, lane = tid & 31;
  const int m0 = blockIdx.x * 128;
  const int n0 = blockIdx.y * 64;

  const int arow = wv * 16 + (lane & 15);
  const int klo = (lane < 16) ? 0 : 16;
  const int kb  = (lane < 16) ? 0 : 16;

  v8f acc[4] = {};
  for (int kc = 0; kc < K; kc += 128) {
    __syncthreads();
    // A chunk: 128x128 bf16 as 16-byte granules (2048 total, 8 per thread).
    #pragma unroll
    for (int i = tid; i < 2048; i += 256) {
      int r = i >> 4, gq = i & 15;
      *reinterpret_cast<uint4*>(&As[r][gq * 8]) =
          *reinterpret_cast<const uint4*>(A + (size_t)(m0 + r) * K + kc + gq * 8);
    }
    // B chunk: 64x128, read float4 along K, store transposed-in-n (row = n).
    #pragma unroll
    for (int i = tid; i < 2048; i += 256) {
      int nn = i >> 5, k4 = i & 31;
      float4 w4 = reinterpret_cast<const float4*>(
          W + (size_t)(n0 + nn) * K + kc)[k4];
      v4bf pk;
      pk[0] = (bf16)w4.x; pk[1] = (bf16)w4.y;
      pk[2] = (bf16)w4.z; pk[3] = (bf16)w4.w;
      *reinterpret_cast<v4bf*>(&Bst[nn][k4 * 4]) = pk;
    }
    __syncthreads();

    #pragma unroll
    for (int ks = 0; ks < 4; ++ks) {
      const int kk = ks * 32;
      v16bf a;   // A fragment (k = klo + i, same order as bm), contiguous reads
      #pragma unroll
      for (int i = 0; i < 8; ++i) {
        a[i]     = As[arow][kk + klo + i];
        a[8 + i] = As[arow][kk + klo + 8 + i];
      }
      #pragma unroll
      for (int j = 0; j < 4; ++j) {
        const int col = j * 16 + (lane & 15);
        v16bf bm;  // B[k][n] = Bst[n][k], contiguous in k
        #pragma unroll
        for (int i = 0; i < 16; ++i) bm[i] = Bst[col][kk + kb + i];
        acc[j] = __builtin_amdgcn_wmma_f32_16x16x32_bf16(
            false, a, false, bm, (short)0, acc[j], false, false);
      }
    }
  }

  // Epilogue (D layout: lane=col, vec r = row r (+8 for upper half)).
  const int rbase = wv * 16 + ((lane < 16) ? 0 : 8);
  #pragma unroll
  for (int j = 0; j < 4; ++j) {
    const int col = n0 + j * 16 + (lane & 15);
    const float bv = bias[col];
    #pragma unroll
    for (int r = 0; r < 8; ++r) {
      const size_t idx = (size_t)(m0 + rbase + r) * N + col;
      float v = acc[j][r] + bv;
      if (EPI == 0) {
        ((bf16*)outv)[idx] = (bf16)v;
      } else if (EPI == 1) {
        float gl = 0.5f * v * (1.0f + erff(v * 0.70710678118654752f));
        ((bf16*)outv)[idx] = (bf16)gl;
      } else if (EPI == 2) {
        ((float*)outv)[idx] = v + aux0[idx] + aux1[idx];
      } else {
        ((float*)outv)[idx] = v + aux0[idx];
      }
    }
  }
}

// ---------------------------------------------------------------------------
// Windowed flash attention, transposed-score form. grid = (NWIN*NHEAD, 4).
// S^T = K*Q^T (lane = query column) -> per-lane scalar softmax state with one
// shfl_xor(16) per reduction; P^T feeds O^T = V^T*P^T via an 8-shuffle
// half-wave exchange. Window token map precomputed once in LDS.
// Window mapping: window (b,wh,ww), token n = d*49 + i*7 + j ->
// spatial h = i*8 + wh, w = j*8 + ww.
// ---------------------------------------------------------------------------
#define QPAD 40   // 32 + 8
__device__ __forceinline__ int win_token(int wi, int n) {
  int b  = wi >> 6;
  int wh = (wi >> 3) & 7;
  int ww = wi & 7;
  int d  = n / 49;
  int rem = n % 49;
  int i  = rem / 7;
  int j  = rem % 7;
  return ((b * 8 + d) * 56 + (i * 8 + wh)) * 56 + (j * 8 + ww);
}

__global__ __launch_bounds__(256) void attn_kernel(
    const bf16* __restrict__ qkv, bf16* __restrict__ outp)
{
  __shared__ bf16 Qs[128][QPAD];   // Q rows (10 KB)
  __shared__ bf16 Ks[32][QPAD];    // K tile, [key][hd]
  __shared__ bf16 Vt[32][QPAD];    // V tile transposed, [hd][key]
  __shared__ int  tokmap[512];     // window token -> flat token (clamped)

  const int tid = threadIdx.x;
  const int wv = tid >> 5, lane = tid & 31;
  const int l15 = lane & 15;
  const bool hi = lane >= 16;
  const int wi = blockIdx.x >> 2;
  const int head = blockIdx.x & 3;
  const int q0 = blockIdx.y * 128;
  const float scale = 0.17677669529663687f; // 1/sqrt(32)

  // Token map (clamped at NTOKW-1) -- div/mod done once per block.
  #pragma unroll
  for (int i = tid; i < 512; i += 256)
    tokmap[i] = win_token(wi, (i < NTOKW) ? i : (NTOKW - 1));
  __syncthreads();

  // Stage Q tile, 16B granules (zeros for padded rows).
  #pragma unroll
  for (int i = tid; i < 512; i += 256) {   // 128 rows x 4 granules
    int r = i >> 2, gq = i & 3;
    int n = q0 + r;
    uint4 val = {0u, 0u, 0u, 0u};
    if (n < NTOKW)
      val = *reinterpret_cast<const uint4*>(
          qkv + (size_t)tokmap[n] * 384 + head * HD + gq * 8);
    *reinterpret_cast<uint4*>(&Qs[r][gq * 8]) = val;
  }
  __syncthreads();

  // Q^T B-fragment (loop invariant): B[k=hd][n=query], contiguous in hd.
  const int qrow = wv * 16 + l15;
  const int kb2 = hi ? 16 : 0;
  v16bf bq;
  #pragma unroll
  for (int i = 0; i < 16; ++i) bq[i] = Qs[qrow][kb2 + i];

  float mrun = -3.0e38f, lrun = 0.0f;   // per-lane = per-query
  v8f ot0 = {}, ot1 = {};               // O^T: hd rows 0-15 / 16-31

  const int klo = hi ? 16 : 0;

  for (int kt = 0; kt < 13; ++kt) {     // ceil(392/32) key tiles
    __syncthreads();
    if (tid < 128) {                    // K tile: 32 rows x 4 x 16B granules
      int r = tid >> 2, gq = tid & 3;
      *reinterpret_cast<uint4*>(&Ks[r][gq * 8]) =
          *reinterpret_cast<const uint4*>(
              qkv + (size_t)tokmap[kt * 32 + r] * 384 + CDIM + head * HD + gq * 8);
    }
    #pragma unroll
    for (int i = tid; i < 1024; i += 256) {  // V tile, transposed into LDS
      int r = i >> 5, c = i & 31;
      Vt[c][r] = qkv[(size_t)tokmap[kt * 32 + r] * 384 + 2 * CDIM + head * HD + c];
    }
    __syncthreads();

    // S^T tiles: A = K rows (16 keys x 32 hd), B = Q^T. One WMMA / 16 keys.
    v16bf ak0, ak1;
    #pragma unroll
    for (int i = 0; i < 8; ++i) {
      ak0[i]     = Ks[l15][klo + i];
      ak0[8 + i] = Ks[l15][klo + 8 + i];
      ak1[i]     = Ks[16 + l15][klo + i];
      ak1[8 + i] = Ks[16 + l15][klo + 8 + i];
    }
    v8f s0 = {}, s1 = {};
    s0 = __builtin_amdgcn_wmma_f32_16x16x32_bf16(false, ak0, false, bq,
                                                 (short)0, s0, false, false);
    s1 = __builtin_amdgcn_wmma_f32_16x16x32_bf16(false, ak1, false, bq,
                                                 (short)0, s1, false, false);

    // Online softmax (per-lane scalars; keys live in the vec dimension).
    const int kbase0 = kt * 32 + (hi ? 8 : 0);       // s0 vec r -> key kbase0+r
    const int kbase1 = kbase0 + 16;                  // s1 vec r -> key kbase1+r
    float p0v[8], p1v[8];
    float tm = -3.0e38f;
    #pragma unroll
    for (int r = 0; r < 8; ++r) {
      float a0 = (kbase0 + r < NTOKW) ? s0[r] * scale : -1.0e30f;
      float a1 = (kbase1 + r < NTOKW) ? s1[r] * scale : -1.0e30f;
      p0v[r] = a0; p1v[r] = a1;
      tm = fmaxf(tm, fmaxf(a0, a1));
    }
    tm = fmaxf(tm, __shfl_xor(tm, 16, 32));
    float mnew = fmaxf(mrun, tm);
    float corr = __expf(mrun - mnew);
    mrun = mnew;
    float ssum = 0.0f;
    #pragma unroll
    for (int r = 0; r < 8; ++r) {
      p0v[r] = __expf(p0v[r] - mnew);
      p1v[r] = __expf(p1v[r] - mnew);
      ssum += p0v[r] + p1v[r];
    }
    ssum += __shfl_xor(ssum, 16, 32);
    lrun = lrun * corr + ssum;
    #pragma unroll
    for (int r = 0; r < 8; ++r) { ot0[r] *= corr; ot1[r] *= corr; }

    // Build P^T B-fragment via half-wave exchange (8 shuffles).
    // lane<16 needs keys 0-15 (own p0 + partner p0); lane>=16 keys 16-31.
    v16bf bp;
    #pragma unroll
    for (int i = 0; i < 8; ++i) {
      float send = hi ? p0v[i] : p1v[i];
      float recv = __shfl_xor(send, 16, 32);
      if (!hi) { bp[i] = (bf16)p0v[i]; bp[8 + i] = (bf16)recv; }
      else     { bp[i] = (bf16)recv;   bp[8 + i] = (bf16)p1v[i]; }
    }

    // O^T += V^T * P^T. A = V^T (16 hd x 32 keys) from transposed Vt.
    v16bf av0, av1;
    #pragma unroll
    for (int i = 0; i < 8; ++i) {
      av0[i]     = Vt[l15][klo + i];
      av0[8 + i] = Vt[l15][klo + 8 + i];
      av1[i]     = Vt[16 + l15][klo + i];
      av1[8 + i] = Vt[16 + l15][klo + 8 + i];
    }
    ot0 = __builtin_amdgcn_wmma_f32_16x16x32_bf16(false, av0, false, bp,
                                                  (short)0, ot0, false, false);
    ot1 = __builtin_amdgcn_wmma_f32_16x16x32_bf16(false, av1, false, bp,
                                                  (short)0, ot1, false, false);
  }

  // Store: lane owns query qrow; ot0 vec r -> hd (hi?8:0)+r, ot1 -> +16.
  const int n = q0 + qrow;
  if (n < NTOKW) {
    const int t = tokmap[n];
    const float inv = 1.0f / lrun;
    size_t base = (size_t)t * CDIM + head * HD + (hi ? 8 : 0);
    v4bf a0, a1, b0, b1;
    #pragma unroll
    for (int r = 0; r < 4; ++r) {
      a0[r] = (bf16)(ot0[r] * inv);
      a1[r] = (bf16)(ot0[4 + r] * inv);
      b0[r] = (bf16)(ot1[r] * inv);
      b1[r] = (bf16)(ot1[4 + r] * inv);
    }
    *reinterpret_cast<v4bf*>(outp + base)          = a0;
    *reinterpret_cast<v4bf*>(outp + base + 4)      = a1;
    *reinterpret_cast<v4bf*>(outp + base + 16)     = b0;
    *reinterpret_cast<v4bf*>(outp + base + 20)     = b1;
  }
}

// ---------------------------------------------------------------------------
// Host-side launch.
// ---------------------------------------------------------------------------
extern "C" void kernel_launch(void* const* d_in, const int* in_sizes, int n_in,
                              void* d_out, int out_size, void* d_ws, size_t ws_size,
                              hipStream_t stream) {
  const float* x       = (const float*)d_in[0];
  const float* norm1_g = (const float*)d_in[1];
  const float* norm1_b = (const float*)d_in[2];
  const float* qkv_w   = (const float*)d_in[3];
  const float* qkv_b   = (const float*)d_in[4];
  const float* proj_w  = (const float*)d_in[5];
  const float* proj_b  = (const float*)d_in[6];
  const float* dw_w    = (const float*)d_in[7];
  const float* dw_b    = (const float*)d_in[8];
  const float* norm2_g = (const float*)d_in[9];
  const float* norm2_b = (const float*)d_in[10];
  const float* fc1_w   = (const float*)d_in[11];
  const float* fc1_b   = (const float*)d_in[12];
  const float* fc2_w   = (const float*)d_in[13];
  const float* fc2_b   = (const float*)d_in[14];

  char* w = (char*)d_ws;
  size_t off = 0;
  auto carve = [&](size_t bytes) {
    void* p = w + off;
    off = (off + bytes + 255) & ~(size_t)255;
    return p;
  };
  float* xn_f   = (float*)carve((size_t)NTOK * CDIM * 4);
  bf16*  xn_bf  = (bf16*) carve((size_t)NTOK * CDIM * 2);
  float* conv   = (float*)carve((size_t)NTOK * CDIM * 4);
  bf16*  qkvb   = (bf16*) carve((size_t)NTOK * 384 * 2);
  bf16*  attnb  = (bf16*) carve((size_t)NTOK * CDIM * 2);
  float* x1     = (float*)carve((size_t)NTOK * CDIM * 4);
  bf16*  xm_bf  = (bf16*) carve((size_t)NTOK * CDIM * 2);
  bf16*  hid_bf = (bf16*) carve((size_t)NTOK * HIDDEN * 2);

  // 1. LN1 -> xn (f32 for conv, bf16 for QKV GEMM)
  ln_kernel<<<NTOK / 8, 256, 0, stream>>>(x, norm1_g, norm1_b, xn_bf, xn_f, NTOK);
  // 2. depthwise conv branch
  dwconv_kernel<<<(NTOK * 32) / 256, 256, 0, stream>>>(xn_f, dw_w, dw_b, conv);
  // 3. QKV projection (natural token order)
  gemm_kernel<0><<<dim3(NTOK / 128, 384 / 64), 256, 0, stream>>>(
      xn_bf, qkv_w, qkv_b, nullptr, nullptr, qkvb, NTOK, 384, CDIM);
  // 4. windowed flash attention
  attn_kernel<<<dim3(NWIN * NHEAD, 4), 256, 0, stream>>>(qkvb, attnb);
  // 5. output projection + conv branch + shortcut -> x1
  gemm_kernel<2><<<dim3(NTOK / 128, CDIM / 64), 256, 0, stream>>>(
      attnb, proj_w, proj_b, conv, x, x1, NTOK, CDIM, CDIM);
  // 6. LN2
  ln_kernel<<<NTOK / 8, 256, 0, stream>>>(x1, norm2_g, norm2_b, xm_bf, nullptr, NTOK);
  // 7. fc1 + GELU
  gemm_kernel<1><<<dim3(NTOK / 128, HIDDEN / 64), 256, 0, stream>>>(
      xm_bf, fc1_w, fc1_b, nullptr, nullptr, hid_bf, NTOK, HIDDEN, CDIM);
  // 8. fc2 + residual -> final output
  gemm_kernel<3><<<dim3(NTOK / 128, CDIM / 64), 256, 0, stream>>>(
      hid_bf, fc2_w, fc2_b, x1, nullptr, d_out, NTOK, CDIM, HIDDEN);
  (void)in_sizes; (void)n_in; (void)out_size; (void)ws_size;
}
